// HGraphConv_21612275433487
// MI455X (gfx1250) — matrix operand order repaired and checked by reading, not compiled, not run on hardware
//
#include <hip/hip_runtime.h>
#include <hip/hip_bf16.h>

// ---------------------------------------------------------------------------
// Fused HGraphConv for gfx1250 (MI455X).
//   precision: bf16 WMMA (v_wmma_f32_16x16x32_bf16), f32 accumulate.
//   key fusion: q+kk == stacked @ (Wq+Wk)^T + (bq+bk)  -> single GEMM.
//   input converted to bf16 ONCE into LDS; all fragments then come from
//   ds_load_b128.  LDS = 163,392 B -> 2 workgroups per 320KB WGP, and
//   __launch_bounds__(256,4) keeps VGPRs <= file/4 so 16 waves fit.
// ---------------------------------------------------------------------------

typedef __attribute__((ext_vector_type(16))) __bf16 v16bf;
typedef __attribute__((ext_vector_type(8)))  __bf16 v8bf;
typedef __attribute__((ext_vector_type(4)))  __bf16 v4bf;
typedef __attribute__((ext_vector_type(8)))  float  v8f;
typedef __attribute__((ext_vector_type(4)))  float  v4f;

#define NJ    25          // joints
#define NB    8192        // batch
#define FD    128         // features
#define BT    4           // batches per workgroup
#define NEGV  -9.0e15f

__device__ __forceinline__ v16bf join16(v8bf lo, v8bf hi) {
  return __builtin_shufflevector(lo, hi, 0,1,2,3,4,5,6,7,8,9,10,11,12,13,14,15);
}
__device__ __forceinline__ v8f wmma_bf16(v16bf a, v16bf b, v8f c) {
  return __builtin_amdgcn_wmma_f32_16x16x32_bf16(
      false, a, false, b, (short)0, c, false, false);
}

// ---------------------------------------------------------------------------
// Prep kernel: builds (in d_ws)
//   Apad    : bf16 [4][32][32]   row-softmax adjacencies, zero padded
//   Wpack   : bf16 [4 hop][4 kt][8 nt][32 lane][16]   B-fragments of W
//   Wqkpack : bf16 [4 kt][8 nt][32 lane][16]          B-fragments of (Wq+Wk)^T
// B-fragment convention: lane <-> N column (nt*16+lane%16), 16 K values in table order (8h+e | 16+8h+e-8).
// ---------------------------------------------------------------------------
__global__ void hgc_prep(const float* __restrict__ W,
                         const float* __restrict__ e0, const float* __restrict__ e1,
                         const float* __restrict__ e2, const float* __restrict__ e3,
                         const int* __restrict__ i0, const int* __restrict__ i1,
                         const int* __restrict__ i2, const int* __restrict__ i3,
                         int n0, int n1, int n2, int n3,
                         const float* __restrict__ Wq, const float* __restrict__ Wk,
                         __bf16* __restrict__ Apad,
                         __bf16* __restrict__ Wpack,
                         __bf16* __restrict__ Wqkpack) {
  int tid = blockIdx.x * blockDim.x + threadIdx.x;
  int stride = gridDim.x * blockDim.x;

  // pack W: value = W[hop][K=kt*32+(8h+e | 16+8h+e-8)][N=nt*16+lane%16], h=lane/16
  for (int x = tid; x < 4*4*8*32*16; x += stride) {
    int e = x & 15, lane = (x >> 4) & 31, nt = (x >> 9) & 7,
        kt = (x >> 12) & 3, hop = x >> 14;
    int K = kt*32 + (e < 8 ? 8*(lane>>4) + e : 16 + 8*(lane>>4) + e - 8), N = nt*16 + (lane & 15);
    Wpack[x] = (__bf16)W[(hop*FD + K)*FD + N];
  }
  // pack (Wq+Wk)^T: value = Wq[N][K] + Wk[N][K]
  for (int x = tid; x < 4*8*32*16; x += stride) {
    int e = x & 15, lane = (x >> 4) & 31, nt = (x >> 9) & 7, kt = x >> 12;
    int K = kt*32 + (e < 8 ? 8*(lane>>4) + e : 16 + 8*(lane>>4) + e - 8), N = nt*16 + (lane & 15);
    Wqkpack[x] = (__bf16)(Wq[N*FD + K] + Wk[N*FD + K]);
  }

  if (blockIdx.x == 0) {
    int t = threadIdx.x;
    if (t < 4*NJ) {                        // one thread per (hop,row)
      int k = t / NJ, i = t % NJ;
      const float* ev = (k==0)?e0:(k==1)?e1:(k==2)?e2:e3;
      const int*   ix = (k==0)?i0:(k==1)?i1:(k==2)?i2:i3;
      int nnz        = (k==0)?n0:(k==1)?n1:(k==2)?n2:n3;
      float row[NJ];
      for (int j = 0; j < NJ; ++j) row[j] = NEGV;
      for (int e = 0; e < nnz; ++e)
        if (ix[2*e] == i) row[ix[2*e+1]] = ev[e];
      float m = row[0];
      for (int j = 1; j < NJ; ++j) m = fmaxf(m, row[j]);
      float s = 0.f;
      for (int j = 0; j < NJ; ++j) { row[j] = expf(row[j] - m); s += row[j]; }
      float inv = 1.f / s;
      __bf16* dst = Apad + k*1024 + i*32;
      for (int j = 0;  j < NJ; ++j) dst[j] = (__bf16)(row[j] * inv);
      for (int j = NJ; j < 32; ++j) dst[j] = (__bf16)0.f;
    } else if (t < 128) {                  // zero pad rows i=25..31
      int u = t - 100, k = u / 7, i = NJ + u % 7;
      __bf16* dst = Apad + k*1024 + i*32;
      for (int j = 0; j < 32; ++j) dst[j] = (__bf16)0.f;
    }
  }
}

// ---------------------------------------------------------------------------
// Main fused kernel: one workgroup = 4 batches, 8 waves (wave32).
// ---------------------------------------------------------------------------
__global__ void __launch_bounds__(256, 4)
hgc_main(const float* __restrict__ input,
         const __bf16* __restrict__ Apad,
         const __bf16* __restrict__ Wpack,
         const __bf16* __restrict__ Wqkpack,
         const float* __restrict__ Wa, const float* __restrict__ ba,
         const float* __restrict__ bq, const float* __restrict__ bk,
         const float* __restrict__ bias,
         float* __restrict__ out) {
  __shared__ __bf16 sIn[100 * FD];          // input bf16 [row=j*4+b][f] 25600 B
  __shared__ __bf16 sH[32 * 512];           // h: [b*128+f][j pad32]    32768 B
  __shared__ __bf16 sStk[NJ * 16 * FD];     // stacked: [i][k*4+b][f]  102400 B
  __shared__ float  sScore[BT * NJ * 4];    // scores -> softmax weights 1600 B
  __shared__ float  sWa[FD], sBqk[FD];      //                           1024 B

  const int b0   = blockIdx.x * BT;
  const int wave = threadIdx.x >> 5;
  const int lane = threadIdx.x & 31;
  const int l16  = lane & 15;
  const int hiH  = lane >> 4;               // 0 / 1

  // ---- init small LDS + zero h pad rows (j = 28..31) ----
  if (threadIdx.x < FD) {
    sWa[threadIdx.x]  = Wa[threadIdx.x];
    sBqk[threadIdx.x] = bq[threadIdx.x] + bk[threadIdx.x];
  }
  for (int x = threadIdx.x; x < 4 * 512; x += 256)
    sH[(x >> 2) * 32 + 28 + (x & 3)] = (__bf16)0.f;

  // ---- stage input -> LDS bf16, ONE conversion per element ----
  {
    const int fb = (threadIdx.x & 31) * 4;
    for (int row = threadIdx.x >> 5; row < BT * NJ; row += 8) {
      int j = row >> 2, bb = row & 3;
      v4f x = *(const v4f*)(input + ((size_t)(b0 + bb) * NJ + j) * FD + fb);
      v4bf y;
      y[0] = (__bf16)x[0]; y[1] = (__bf16)x[1];
      y[2] = (__bf16)x[2]; y[3] = (__bf16)x[3];
      *(v4bf*)&sIn[row * FD + fb] = y;
    }
  }
  __syncthreads();

  // =================== per-hop: h = input@W[k];  stacked = A_k@h ==========
  for (int k = 0; k < 4; ++k) {
    // -- stage 1: 7 M-tiles x 8 N-tiles, K=128 (4 steps of 32) --
    // job loop strides by 8 -> nt fixed per wave: B-fragments stay in VGPRs.
    for (int job = wave; job < 56; job += 8) {
      const int m = job >> 3, nt = job & 7;
      int row = m * 16 + l16;
      if (row > BT * NJ - 1) row = BT * NJ - 1;   // clamp pad rows (A*0 later)
      const __bf16* rp = &sIn[row * FD];
      const __bf16* wq0 = Wpack + (size_t)(((k*4)*8 + nt)*32 + lane) * 16;
      __builtin_prefetch(wq0, 0, 0);

      v8f acc; for (int r = 0; r < 8; ++r) acc[r] = 0.f;
      for (int kt = 0; kt < 4; ++kt) {
        v16bf a = join16(*(const v8bf*)(rp + kt*32 + (hiH ? 8 : 0)),
                         *(const v8bf*)(rp + kt*32 + (hiH ? 24 : 16)));
        const __bf16* wp = Wpack + (size_t)((((k*4 + kt)*8) + nt)*32 + lane) * 16;
        v16bf b = join16(*(const v8bf*)wp, *(const v8bf*)(wp + 8));
        acc = wmma_bf16(a, b, acc);
      }
      // write C tile transposed into sH[b*128+f][j] (bf16)
      int f = nt * 16 + l16;
      for (int r = 0; r < 8; ++r) {
        int row2 = m * 16 + r + (hiH ? 8 : 0);       // < 112 -> j2 < 28
        int j2 = row2 >> 2, b2 = row2 & 3;
        sH[(b2 * FD + f) * 32 + j2] = (__bf16)acc[r];
      }
    }
    __syncthreads();

    // -- stage 3: stacked[i][.] = A_k[i][j] @ h[j][.], K=32 one WMMA --
    for (int job = wave; job < 64; job += 8) {
      const int m = job & 1, nt = job >> 1;          // m: i-tile, nt: 512/16
      const __bf16* ap = Apad + k * 1024 + (m * 16 + l16) * 32;
      v16bf a = join16(*(const v8bf*)(ap + (hiH ? 8 : 0)),
                       *(const v8bf*)(ap + (hiH ? 24 : 16)));
      const __bf16* hp = &sH[(nt * 16 + l16) * 32];  // lane <-> column, elements over j
      v16bf b = join16(*(const v8bf*)(hp + (hiH ? 8 : 0)),
                       *(const v8bf*)(hp + (hiH ? 24 : 16)));
      v8f acc; for (int r = 0; r < 8; ++r) acc[r] = 0.f;
      acc = wmma_bf16(a, b, acc);

      int c = nt * 16 + l16, b2 = c >> 7, f = c & 127;
      for (int r = 0; r < 8; ++r) {
        int i = m * 16 + r + (hiH ? 8 : 0);
        if (i < NJ) sStk[(i * 16 + k * 4 + b2) * FD + f] = (__bf16)acc[r];
      }
    }
    __syncthreads();
  }

  // ========== stage 4: scores = Wa . tanh(stacked@(Wq+Wk)^T + bqk) + ba ====
  for (int job = wave; job < NJ; job += 8) {
    const int i = job;                     // 16 rows (k*4+b) = one M-tile
    v16bf afr[4];
    const __bf16* sp = &sStk[(i * 16 + l16) * FD];
    for (int kt = 0; kt < 4; ++kt) {
      int oA = kt*32 + (hiH ? 8 : 0), oB = kt*32 + (hiH ? 24 : 16);
      afr[kt] = join16(*(const v8bf*)(sp + oA), *(const v8bf*)(sp + oB));
    }
    float sacc[8]; for (int r = 0; r < 8; ++r) sacc[r] = 0.f;
    for (int nt = 0; nt < 8; ++nt) {
      float binit = sBqk[nt * 16 + l16];
      v8f acc; for (int r = 0; r < 8; ++r) acc[r] = binit;
      for (int kt = 0; kt < 4; ++kt) {
        const __bf16* wp = Wqkpack + (size_t)(((kt*8) + nt)*32 + lane) * 16;
        v16bf b = join16(*(const v8bf*)wp, *(const v8bf*)(wp + 8));
        acc = wmma_bf16(afr[kt], b, acc);
      }
      float wa = sWa[nt * 16 + l16];
      for (int r = 0; r < 8; ++r) sacc[r] += tanhf(acc[r]) * wa;
    }
    float bav = ba[0];
    for (int r = 0; r < 8; ++r) {          // reduce over the 16 N lanes
      float v = sacc[r];
      v += __shfl_xor(v, 1, 16);
      v += __shfl_xor(v, 2, 16);
      v += __shfl_xor(v, 4, 16);
      v += __shfl_xor(v, 8, 16);
      if (l16 == 0) {
        int M = r + (hiH ? 8 : 0);         // M = k*4 + b
        sScore[(M & 3) * (NJ*4) + i * 4 + (M >> 2)] = v + bav;
      }
    }
  }
  __syncthreads();

  // ========== softmax over joints (axis=1) per (b,k) =======================
  if (threadIdx.x < BT * 4) {
    int b2 = threadIdx.x >> 2, kk = threadIdx.x & 3;
    float* s = &sScore[b2 * (NJ*4) + kk];
    float m = -1e30f;
    for (int i = 0; i < NJ; ++i) m = fmaxf(m, s[i*4]);
    float sum = 0.f;
    for (int i = 0; i < NJ; ++i) sum += expf(s[i*4] - m);
    float inv = 1.f / sum;
    for (int i = 0; i < NJ; ++i) s[i*4] = expf(s[i*4] - m) * inv;
  }
  __syncthreads();

  // ========== stage 7: attended = sum_k stacked*w + bias -> out ============
  const int fb = (threadIdx.x & 31) * 4;
  for (int row = threadIdx.x >> 5; row < BT * NJ; row += 8) {
    int b2 = row / NJ, i = row % NJ;
    v4f acc = *(const v4f*)(bias + fb);    // L2-resident, no LDS needed
    for (int k = 0; k < 4; ++k) {
      float wk = sScore[b2 * (NJ*4) + i * 4 + k];
      v4bf sv = *(const v4bf*)&sStk[(i * 16 + k * 4 + b2) * FD + fb];
      acc[0] += wk * (float)sv[0];
      acc[1] += wk * (float)sv[1];
      acc[2] += wk * (float)sv[2];
      acc[3] += wk * (float)sv[3];
    }
    *(v4f*)(out + ((size_t)(b0 + b2) * NJ + i) * FD + fb) = acc;
  }
}

// ---------------------------------------------------------------------------
extern "C" void kernel_launch(void* const* d_in, const int* in_sizes, int n_in,
                              void* d_out, int out_size, void* d_ws, size_t ws_size,
                              hipStream_t stream) {
  const float* input = (const float*)d_in[0];
  const float* W     = (const float*)d_in[1];
  const float* e0    = (const float*)d_in[2];
  const float* e1    = (const float*)d_in[3];
  const float* e2    = (const float*)d_in[4];
  const float* e3    = (const float*)d_in[5];
  const float* Wq    = (const float*)d_in[6];
  const float* bq    = (const float*)d_in[7];
  const float* Wk    = (const float*)d_in[8];
  const float* bk    = (const float*)d_in[9];
  const float* Wa    = (const float*)d_in[10];
  const float* ba    = (const float*)d_in[11];
  const float* bias  = (const float*)d_in[12];
  const int*   idx0  = (const int*)d_in[13];
  const int*   idx1  = (const int*)d_in[14];
  const int*   idx2  = (const int*)d_in[15];
  const int*   idx3  = (const int*)d_in[16];

  // workspace carve (bf16 packs)
  __bf16* Apad    = (__bf16*)d_ws;                             //   8 KB
  __bf16* Wpack   = (__bf16*)((char*)d_ws + 8192);             // 128 KB
  __bf16* Wqkpack = (__bf16*)((char*)d_ws + 8192 + 131072);    //  32 KB

  hgc_prep<<<64, 256, 0, stream>>>(W, e0, e1, e2, e3, idx0, idx1, idx2, idx3,
                                   in_sizes[2], in_sizes[3], in_sizes[4], in_sizes[5],
                                   Wq, Wk, Apad, Wpack, Wqkpack);

  hgc_main<<<NB / BT, 256, 0, stream>>>(input, Apad, Wpack, Wqkpack,
                                        Wa, ba, bq, bk, bias, (float*)d_out);
}
